// Attention_Module_403726926153
// MI455X (gfx1250) — matrix operand order repaired and checked by reading, not compiled, not run on hardware
//
#include <hip/hip_runtime.h>

#define C_DIM 256
#define T_DIM 4096

typedef __attribute__((ext_vector_type(16))) __bf16 v16bf;
typedef __attribute__((ext_vector_type(8)))  float  v8f;

struct U32x8 { uint4 lo, hi; };

__device__ __forceinline__ v16bf make_frag(uint4 lo, uint4 hi) {
  U32x8 t{lo, hi};
  return __builtin_bit_cast(v16bf, t);
}

// native f32 -> bf16 conversion (backend picks v_cvt_*_bf16_f32), packed pair
__device__ __forceinline__ unsigned pk_bf16x2(float a, float b) {
  unsigned short ua = __builtin_bit_cast(unsigned short, (__bf16)a);
  unsigned short ub = __builtin_bit_cast(unsigned short, (__bf16)b);
  return (unsigned)ua | ((unsigned)ub << 16);
}

__device__ __forceinline__ unsigned short bf16r(float a) {
  return __builtin_bit_cast(unsigned short, (__bf16)a);
}

#if __has_builtin(__builtin_amdgcn_exp2f)
#define EXP2F(x) __builtin_amdgcn_exp2f(x)
#else
#define EXP2F(x) exp2f(x)
#endif

// -------------------------------------------------------------------------
// Pass 1: energy = x x^T (64x256 slab per block, K=T), fused softmax of
// (rowmax - energy) == exp(rowmin - energy)/sum, attention stored bf16 in ws.
// Block: 256 threads (8 waves). Wave w: M-tile (w&3), N-tiles (w>>2)*8..+8.
// Double-buffered LDS staging; pipelined B-fragment loads.
// -------------------------------------------------------------------------
__global__ __launch_bounds__(256)
void attn_energy_softmax(const float* __restrict__ x,
                         unsigned short* __restrict__ attn) {
  __shared__ unsigned Xs[2][C_DIM * 16];  // 2 x (256 rows x 32 bf16) = 32 KB
  __shared__ float    wred[8 * 16];

  const int b    = blockIdx.y;
  const int i0   = blockIdx.x * 64;
  const int tid  = threadIdx.x;
  const int wave = tid >> 5, lane = tid & 31;
  const int mt   = wave & 3, ntb = (wave >> 2) * 8;
  const int half = lane >> 4, lr = lane & 15;

  const float* xb  = x + (size_t)b * C_DIM * T_DIM;
  const float* src = xb + (size_t)tid * T_DIM;   // one x-row per thread

  v8f acc[8] = {};

  const int arow = i0 + mt * 16 + lr;

  // stage chunk 0 into Xs[0]
#pragma unroll
  for (int q = 0; q < 4; ++q) {
    float4 va = *(const float4*)(src + q * 8);
    float4 vb = *(const float4*)(src + q * 8 + 4);
    uint4 o;
    o.x = pk_bf16x2(va.x, va.y);
    o.y = pk_bf16x2(va.z, va.w);
    o.z = pk_bf16x2(vb.x, vb.y);
    o.w = pk_bf16x2(vb.z, vb.w);
    *(uint4*)&Xs[0][tid * 16 + q * 4] = o;
  }

  for (int kt = 0; kt < T_DIM; kt += 32) {
    const int cur = (kt >> 5) & 1;
    __syncthreads();   // staging of Xs[cur] complete

    // stage next chunk into the other buffer (overlaps with WMMAs below)
    if (kt + 32 < T_DIM) {
      const float* sn = src + kt + 32;
#pragma unroll
      for (int q = 0; q < 4; ++q) {
        float4 va = *(const float4*)(sn + q * 8);
        float4 vb = *(const float4*)(sn + q * 8 + 4);
        uint4 o;
        o.x = pk_bf16x2(va.x, va.y);
        o.y = pk_bf16x2(va.z, va.w);
        o.z = pk_bf16x2(vb.x, vb.y);
        o.w = pk_bf16x2(vb.z, vb.w);
        *(uint4*)&Xs[cur ^ 1][tid * 16 + q * 4] = o;
      }
      if (kt + 64 < T_DIM) __builtin_prefetch(src + kt + 64, 0, 0);
    }

    const unsigned* Xc = Xs[cur];
    // A frag: 16x32 bf16; lane half 0 -> K {0..15}, half 1 -> {16..31}
    v16bf afrag = make_frag(*(const uint4*)&Xc[arow * 16 + 8 * half],
                            *(const uint4*)&Xc[arow * 16 + 8 * half + 4]);
    // pipelined B frags: load nt+1 before WMMA(nt)
    v16bf bfrag = make_frag(*(const uint4*)&Xc[(ntb * 16 + lr) * 16 + 8 * half],
                            *(const uint4*)&Xc[(ntb * 16 + lr) * 16 + 8 * half + 4]);
#pragma unroll
    for (int nt = 0; nt < 8; ++nt) {
      v16bf bnext;
      if (nt < 7) {
        const int brow = (ntb + nt + 1) * 16 + lr;
        bnext = make_frag(*(const uint4*)&Xc[brow * 16 + 8 * half],
                          *(const uint4*)&Xc[brow * 16 + 8 * half + 4]);
      }
      acc[nt] = __builtin_amdgcn_wmma_f32_16x16x32_bf16(
          false, afrag, false, bfrag, (short)0, acc[nt], false, false);
      if (nt < 7) bfrag = bnext;
    }
  }

  // ---- row-min over all 256 j (C layout: VGPR r -> M=r+8*half, N=lane&15) ----
  float rmin[8];
#pragma unroll
  for (int r = 0; r < 8; ++r) {
    float m = acc[0][r];
#pragma unroll
    for (int nt = 1; nt < 8; ++nt) m = fminf(m, acc[nt][r]);
#pragma unroll
    for (int s = 1; s < 16; s <<= 1) m = fminf(m, __shfl_xor(m, s, 16));
    rmin[r] = m;
  }
  if (lr == 0) {
#pragma unroll
    for (int r = 0; r < 8; ++r) wred[wave * 16 + 8 * half + r] = rmin[r];
  }
  __syncthreads();
  float rowmin[8];
#pragma unroll
  for (int r = 0; r < 8; ++r)
    rowmin[r] = fminf(wred[wave * 16 + 8 * half + r],
                      wred[(wave ^ 4) * 16 + 8 * half + r]);
  __syncthreads();

  // ---- p = exp(rowmin - e); row sums ----
  float rsum[8];
#pragma unroll
  for (int r = 0; r < 8; ++r) {
    float s = 0.f;
#pragma unroll
    for (int nt = 0; nt < 8; ++nt) {
      float p = EXP2F((rowmin[r] - acc[nt][r]) * 1.4426950408889634f);
      acc[nt][r] = p;
      s += p;
    }
#pragma unroll
    for (int sh = 1; sh < 16; sh <<= 1) s += __shfl_xor(s, sh, 16);
    rsum[r] = s;
  }
  if (lr == 0) {
#pragma unroll
    for (int r = 0; r < 8; ++r) wred[wave * 16 + 8 * half + r] = rsum[r];
  }
  __syncthreads();

  unsigned short* ab = attn + (size_t)b * C_DIM * C_DIM;
#pragma unroll
  for (int r = 0; r < 8; ++r) {
    const float inv = 1.0f / (wred[wave * 16 + 8 * half + r] +
                              wred[(wave ^ 4) * 16 + 8 * half + r]);
    const int i = i0 + mt * 16 + r + 8 * half;
#pragma unroll
    for (int nt = 0; nt < 8; ++nt) {
      const int j = (ntb + nt) * 16 + lr;
      ab[(size_t)i * C_DIM + j] = bf16r(acc[nt][r] * inv);
    }
  }
}

// -------------------------------------------------------------------------
// Pass 2: out = gamma * (attn @ x) + x. Block computes 64 i-rows x 256 t-cols.
// B operand needs x^T over j: stage x chunk transposed [t][j] in LDS (pitch 48
// bf16 keeps ds_load_b128 frags 16B-aligned). A frags load bf16 attn directly
// from global (L2-resident, 4 MB). Double-buffered staging, pipelined frags.
// Output stores use the NT temporal hint: out is write-once, keep L2 for x.
// -------------------------------------------------------------------------
__global__ __launch_bounds__(256)
void attn_apply(const float* __restrict__ x,
                const unsigned short* __restrict__ attn,
                const float* __restrict__ gamma,
                float* __restrict__ out) {
  __shared__ unsigned Bs[2][C_DIM * 24];  // 2 x (256 t-rows x 48 bf16) = 48 KB

  const int b    = blockIdx.z;
  const int i0   = blockIdx.y * 64;
  const int t0   = blockIdx.x * 256;
  const int tid  = threadIdx.x;
  const int wave = tid >> 5, lane = tid & 31;
  const int mt   = wave & 3, ntb = (wave >> 2) * 8;
  const int half = lane >> 4, lr = lane & 15;

  const float* xb = x + (size_t)b * C_DIM * T_DIM;
  const unsigned short* ap =
      attn + (size_t)b * C_DIM * C_DIM + (size_t)(i0 + mt * 16 + lr) * C_DIM;

  v8f acc[8] = {};

  const int jpair = tid >> 4;   // 0..15 -> j rows (2*jpair, 2*jpair+1)
  const int tpart = tid & 15;   // 0..15 -> 16 t-values each

  // stage chunk j0=0 into Bs[0]
  {
    const float* srcA = xb + (size_t)(2 * jpair) * T_DIM + t0 + tpart * 16;
    const float* srcB = srcA + T_DIM;
#pragma unroll
    for (int q = 0; q < 4; ++q) {
      float4 a = *(const float4*)(srcA + q * 4);
      float4 c = *(const float4*)(srcB + q * 4);
      const int tl = tpart * 16 + q * 4;
      Bs[0][(tl + 0) * 24 + jpair] = pk_bf16x2(a.x, c.x);
      Bs[0][(tl + 1) * 24 + jpair] = pk_bf16x2(a.y, c.y);
      Bs[0][(tl + 2) * 24 + jpair] = pk_bf16x2(a.z, c.z);
      Bs[0][(tl + 3) * 24 + jpair] = pk_bf16x2(a.w, c.w);
    }
  }

  for (int j0 = 0; j0 < C_DIM; j0 += 32) {
    const int cur = (j0 >> 5) & 1;
    __syncthreads();   // staging of Bs[cur] complete

    if (j0 + 32 < C_DIM) {
      const float* srcA = xb + (size_t)(j0 + 32 + 2 * jpair) * T_DIM + t0 + tpart * 16;
      const float* srcB = srcA + T_DIM;
#pragma unroll
      for (int q = 0; q < 4; ++q) {
        float4 a = *(const float4*)(srcA + q * 4);
        float4 c = *(const float4*)(srcB + q * 4);
        const int tl = tpart * 16 + q * 4;
        Bs[cur ^ 1][(tl + 0) * 24 + jpair] = pk_bf16x2(a.x, c.x);
        Bs[cur ^ 1][(tl + 1) * 24 + jpair] = pk_bf16x2(a.y, c.y);
        Bs[cur ^ 1][(tl + 2) * 24 + jpair] = pk_bf16x2(a.z, c.z);
        Bs[cur ^ 1][(tl + 3) * 24 + jpair] = pk_bf16x2(a.w, c.w);
      }
    }

    const unsigned* Bc = Bs[cur];
    // A frag from global bf16 attention (row-major, K=j contiguous)
    v16bf afrag = make_frag(*(const uint4*)(ap + j0 + 16 * half),
                            *(const uint4*)(ap + j0 + 16 * half + 8));
    v16bf bfrag = make_frag(*(const uint4*)&Bc[(ntb * 16 + lr) * 24 + 8 * half],
                            *(const uint4*)&Bc[(ntb * 16 + lr) * 24 + 8 * half + 4]);
#pragma unroll
    for (int nt = 0; nt < 8; ++nt) {
      v16bf bnext;
      if (nt < 7) {
        const int n = (ntb + nt + 1) * 16 + lr;
        bnext = make_frag(*(const uint4*)&Bc[n * 24 + 8 * half],
                          *(const uint4*)&Bc[n * 24 + 8 * half + 4]);
      }
      acc[nt] = __builtin_amdgcn_wmma_f32_16x16x32_bf16(
          false, afrag, false, bfrag, (short)0, acc[nt], false, false);
      if (nt < 7) bfrag = bnext;
    }
  }

  const float g = gamma[0];
  float* ob = out + (size_t)b * C_DIM * T_DIM;
#pragma unroll
  for (int r = 0; r < 8; ++r) {
    const int i = i0 + mt * 16 + r + 8 * half;
#pragma unroll
    for (int nt = 0; nt < 8; ++nt) {
      const int t = t0 + (ntb + nt) * 16 + lr;
      const size_t idx = (size_t)i * T_DIM + t;
      // NT store: out is write-once; don't evict L2-resident x/attn
      __builtin_nontemporal_store(g * acc[nt][r] + xb[idx], &ob[idx]);
    }
  }
}

extern "C" void kernel_launch(void* const* d_in, const int* in_sizes, int n_in,
                              void* d_out, int out_size, void* d_ws, size_t ws_size,
                              hipStream_t stream) {
  const float* x     = (const float*)d_in[0];
  const float* gamma = (const float*)d_in[1];
  float* out = (float*)d_out;
  const int Bn = in_sizes[0] / (C_DIM * T_DIM);   // 32

  unsigned short* attn = (unsigned short*)d_ws;   // B*C*C bf16 = 4 MB scratch

  attn_energy_softmax<<<dim3(C_DIM / 64, Bn), 256, 0, stream>>>(x, attn);
  attn_apply<<<dim3(T_DIM / 256, C_DIM / 64, Bn), 256, 0, stream>>>(x, attn, gamma, out);
}
